// DevignModel_82343112999443
// MI455X (gfx1250) — compile-verified, matrix operand order checked by reading, not run on hardware
//
#include <hip/hip_runtime.h>
#include <hip/hip_bf16.h>
#include <math.h>

// Model dims (fixed by reference)
#define D_IN 128
#define DD   64
#define HH   4
#define DK   16
#define LL   4
#define TT   3
#define RR   32

typedef __attribute__((ext_vector_type(16))) _Float16 v16h;
typedef __attribute__((ext_vector_type(8)))  _Float16 v8h;
typedef __attribute__((ext_vector_type(8)))  float    v8f;

// etype -> src/dst node type tables (from reference's hardcoded ranges)
__constant__ int c_src_nt[RR] = {
  0,0,0,0,0,0,0,0,0,0,
  1,1,1,1,1,1,1,1,1,1,1,1,
  2,2,2,2,2,2,2,2,2,2};
__constant__ int c_dst_nt[RR] = {
  0,0,0, 1,1,1,1, 2,2,2,
  0,0,0,0, 1,1,1,1, 2,2,2,2,
  0,0,0, 1,1,1,1, 2,2,2};

__device__ __forceinline__ float atomicMaxFloat(float* addr, float value) {
  if (value >= 0.0f)
    return __int_as_float(atomicMax((int*)addr, __float_as_int(value)));
  else
    return __uint_as_float(atomicMin((unsigned int*)addr, __float_as_uint(value)));
}

__device__ __forceinline__ float sigmoidf_(float x) {
  return 1.0f / (1.0f + __expf(-x));
}

// ---------------------------------------------------------------------------
// Weight pre-conversion: f32 [T,DIN,64] -> f16 transposed [T,64,DIN]
// so B fragments load as two contiguous 16B chunks per k-step.
// ---------------------------------------------------------------------------
__global__ void convert_w_kernel(const float* __restrict__ W,
                                 _Float16* __restrict__ Wt,
                                 int Tn, int DIN) {
  int idx = blockIdx.x * blockDim.x + threadIdx.x;
  int total = Tn * DIN * DD;
  if (idx >= total) return;
  int t   = idx / (DIN * DD);
  int rem = idx - t * (DIN * DD);
  int col = rem / DIN;
  int K   = rem - col * DIN;
  Wt[idx] = (_Float16)W[((size_t)t * DIN + K) * DD + col];
}

// ---------------------------------------------------------------------------
// WMMA per-node-type linear: y[n,64] = x[n,DIN] @ W[ntype[n]] + b[ntype[n]]
// One wave handles a tile of 16 nodes; computes all T=3 types with
// v_wmma_f32_16x16x32_f16 and stores only rows whose ntype matches.
// A-fragment (16-bit A 16x32): lane L holds row M=L&15; half i holds
//   K = (i<8?0:16) + (L>=16?8:0) + (i&7).  B symmetric with lane=col,
//   fed from the [t][col][K] pre-transposed f16 weights (2x b128 per frag).
// ---------------------------------------------------------------------------
template<int DIN, int ACT>
__global__ void proj_kernel(const float* __restrict__ x,
                            const _Float16* __restrict__ Wt, // [T,64,DIN] f16
                            const float* __restrict__ bias,  // [T,64]
                            const int*   __restrict__ ntype,
                            float* __restrict__ y, int n) {
  const int wave  = (blockIdx.x * blockDim.x + threadIdx.x) >> 5;
  const int lane  = threadIdx.x & 31;
  const int node0 = wave * 16;
  if (node0 >= n) return;
  const int row    = lane & 15;
  const int hiHalf = (lane >> 4) & 1;
  constexpr int KSTEPS = DIN / 32;

  // Load A fragments (shared across all col-tiles and types): 4x float4 each
  v16h aF[KSTEPS];
  const int m_node = node0 + row;
  const float* xr = x + (size_t)m_node * DIN;
  for (int ks = 0; ks < KSTEPS; ++ks) {
    v16h a;
    if (m_node < n) {
      const float* xp = xr + ks * 32 + hiHalf * 8;
      float4 a0 = *(const float4*)(xp);
      float4 a1 = *(const float4*)(xp + 4);
      float4 a2 = *(const float4*)(xp + 16);
      float4 a3 = *(const float4*)(xp + 20);
      a[0] = (_Float16)a0.x; a[1] = (_Float16)a0.y;
      a[2] = (_Float16)a0.z; a[3] = (_Float16)a0.w;
      a[4] = (_Float16)a1.x; a[5] = (_Float16)a1.y;
      a[6] = (_Float16)a1.z; a[7] = (_Float16)a1.w;
      a[8]  = (_Float16)a2.x; a[9]  = (_Float16)a2.y;
      a[10] = (_Float16)a2.z; a[11] = (_Float16)a2.w;
      a[12] = (_Float16)a3.x; a[13] = (_Float16)a3.y;
      a[14] = (_Float16)a3.z; a[15] = (_Float16)a3.w;
    } else {
#pragma unroll
      for (int i = 0; i < 16; ++i) a[i] = (_Float16)0.0f;
    }
    aF[ks] = a;
  }

  for (int ct = 0; ct < 4; ++ct) {
    const int col = ct * 16 + row;           // lane&15 doubles as column index
    for (int t = 0; t < TT; ++t) {
      v8f c = {};
      const _Float16* wcol = Wt + ((size_t)t * DD + col) * DIN + hiHalf * 8;
      for (int ks = 0; ks < KSTEPS; ++ks) {
        const _Float16* wp = wcol + ks * 32;
        v8h lo = *(const v8h*)(wp);
        v8h hi = *(const v8h*)(wp + 16);
        v16h b;
#pragma unroll
        for (int i = 0; i < 8; ++i) { b[i] = lo[i]; b[8 + i] = hi[i]; }
        c = __builtin_amdgcn_wmma_f32_16x16x32_f16(
              false, aF[ks], false, b, (short)0, c, false, false);
      }
      const float bv = bias[t * DD + col];
#pragma unroll
      for (int r = 0; r < 8; ++r) {
        int m = node0 + r + hiHalf * 8;
        if (m < n && ntype[m] == t) {
          float val = c[r] + bv;
          if (ACT == 1) val = tanhf(val);
          y[(size_t)m * DD + col] = val;
        }
      }
    }
  }
}

// ---------------------------------------------------------------------------
// Fill kernel for scratch init
// ---------------------------------------------------------------------------
__global__ void fill_kernel(float* __restrict__ p, float v, size_t n) {
  size_t i = (size_t)blockIdx.x * blockDim.x + threadIdx.x;
  if (i < n) p[i] = v;
}

// ---------------------------------------------------------------------------
// Edge pass 1: logits + segment max + segment count.
// grid = (ceil(E/256), H). LDS caches rel_att[l,:,h,:,:] (32KB, stride-257
// padded to dodge bank conflicts on per-lane etype).
// ---------------------------------------------------------------------------
__global__ void edge_logits_kernel(const int* __restrict__ src,
                                   const int* __restrict__ dst,
                                   const int* __restrict__ et,
                                   const float* __restrict__ k,
                                   const float* __restrict__ q,
                                   const float* __restrict__ rel_att_l, // [R,H,16,16]
                                   const float* __restrict__ rel_pri_l, // [R,H]
                                   const float* __restrict__ nta_l,     // [T]
                                   const float* __restrict__ nta1_l,    // [T]
                                   const float* __restrict__ afc_l,     // [32]
                                   const float* __restrict__ wgt_l,     // [1]
                                   float* __restrict__ logits,          // [E,H]
                                   float* __restrict__ segmax,          // [N*R,H]
                                   float* __restrict__ cnt,             // [N*R]
                                   int E) {
  __shared__ float s_rel[RR * 257];
  const int h = blockIdx.y;
  for (int idx = threadIdx.x; idx < RR * 256; idx += blockDim.x) {
    int r = idx >> 8, o = idx & 255;
    s_rel[r * 257 + o] = rel_att_l[((size_t)r * HH + h) * 256 + o];
  }
  __syncthreads();

  const int e = blockIdx.x * blockDim.x + threadIdx.x;
  if (e >= E) return;
  const int s = src[e], d = dst[e], r = et[e];

  const float* kp = k + (size_t)s * DD + h * DK;
  const float* qp = q + (size_t)d * DD + h * DK;
  float kv[DK], qv[DK];
#pragma unroll
  for (int i = 0; i < DK; i += 4) {
    float4 kt = *(const float4*)(kp + i);
    float4 qt = *(const float4*)(qp + i);
    kv[i] = kt.x; kv[i+1] = kt.y; kv[i+2] = kt.z; kv[i+3] = kt.w;
    qv[i] = qt.x; qv[i+1] = qt.y; qv[i+2] = qt.z; qv[i+3] = qt.w;
  }

  const float* m = &s_rel[r * 257];
  float att = 0.0f;
#pragma unroll 4
  for (int f = 0; f < DK; ++f) {
    float key = 0.0f;
#pragma unroll
    for (int d2 = 0; d2 < DK; ++d2) key = fmaf(kv[d2], m[d2 * DK + f], key);
    att = fmaf(qv[f], key, att);
  }
  att *= rel_pri_l[r * HH + h] * 0.25f;   // / sqrt(DK)

  const float asrc = nta_l[c_src_nt[r]];
  const float adst = nta1_l[c_dst_nt[r]];
  float na = 0.0f;
#pragma unroll
  for (int f = 0; f < DK; ++f)
    na += adst * qv[f] * afc_l[f] + asrc * kv[f] * afc_l[DK + f];
  na = (na > 0.0f) ? na : 0.01f * na;     // leaky_relu

  const float lg = att + sigmoidf_(wgt_l[0]) * na;
  logits[(size_t)e * HH + h] = lg;
  atomicMaxFloat(&segmax[((size_t)d * RR + r) * HH + h], lg);
  if (h == 0) atomicAdd(&cnt[(size_t)d * RR + r], 1.0f);
}

// ---------------------------------------------------------------------------
// Edge pass 2: exp(logit - segmax), store in place, accumulate denominator.
// ---------------------------------------------------------------------------
__global__ void edge_exp_kernel(const int* __restrict__ dst,
                                const int* __restrict__ et,
                                float* __restrict__ logits,
                                const float* __restrict__ segmax,
                                float* __restrict__ den, int E) {
  int i = blockIdx.x * blockDim.x + threadIdx.x;
  if (i >= E * HH) return;
  int e = i >> 2, h = i & 3;
  int d = dst[e], r = et[e];
  size_t seg = ((size_t)d * RR + r) * HH + h;
  float ex = __expf(logits[i] - segmax[seg]);
  logits[i] = ex;
  atomicAdd(&den[seg], ex);
}

// ---------------------------------------------------------------------------
// Edge pass 3: p = ex/den; val = v[src] @ rel_msg[etype,h]; scatter-add
// p*val into per-node accumulator (summing over etypes directly).
// ---------------------------------------------------------------------------
__global__ void edge_accum_kernel(const int* __restrict__ src,
                                  const int* __restrict__ dst,
                                  const int* __restrict__ et,
                                  const float* __restrict__ v,
                                  const float* __restrict__ rel_msg_l, // [R,H,16,16]
                                  const float* __restrict__ logits,    // holds ex
                                  const float* __restrict__ den,
                                  float* __restrict__ acc,             // [N,64]
                                  int E) {
  __shared__ float s_rel[RR * 257];
  const int h = blockIdx.y;
  for (int idx = threadIdx.x; idx < RR * 256; idx += blockDim.x) {
    int r = idx >> 8, o = idx & 255;
    s_rel[r * 257 + o] = rel_msg_l[((size_t)r * HH + h) * 256 + o];
  }
  __syncthreads();

  const int e = blockIdx.x * blockDim.x + threadIdx.x;
  if (e >= E) return;
  const int s = src[e], d = dst[e], r = et[e];

  const float p = logits[(size_t)e * HH + h] /
                  den[((size_t)d * RR + r) * HH + h];
  const float* vp = v + (size_t)s * DD + h * DK;
  float vv[DK];
#pragma unroll
  for (int i = 0; i < DK; i += 4) {
    float4 vt = *(const float4*)(vp + i);
    vv[i] = vt.x; vv[i+1] = vt.y; vv[i+2] = vt.z; vv[i+3] = vt.w;
  }

  const float* m = &s_rel[r * 257];
  float* ap = acc + (size_t)d * DD + h * DK;
#pragma unroll 4
  for (int f = 0; f < DK; ++f) {
    float val = 0.0f;
#pragma unroll
    for (int d2 = 0; d2 < DK; ++d2) val = fmaf(vv[d2], m[d2 * DK + f], val);
    atomicAdd(&ap[f], p * val);
  }
}

// ---------------------------------------------------------------------------
// Divide node accumulator by number of present etypes (cross-etype mean).
// ---------------------------------------------------------------------------
__global__ void node_div_kernel(float* __restrict__ acc,
                                const float* __restrict__ cnt, int n) {
  int nd = blockIdx.x * blockDim.x + threadIdx.x;
  if (nd >= n) return;
  float c = 0.0f;
  for (int r = 0; r < RR; ++r)
    c += (cnt[(size_t)nd * RR + r] > 0.0f) ? 1.0f : 0.0f;
  const float inv = 1.0f / fmaxf(c, 1.0f);
  float* ap = acc + (size_t)nd * DD;
  for (int i = 0; i < DD; ++i) ap[i] *= inv;
}

// ---------------------------------------------------------------------------
// Epilogue: skip-blend + layernorm per node.
// ---------------------------------------------------------------------------
__global__ void epilogue_kernel(const float* __restrict__ trans,
                                const float* __restrict__ x,
                                const int* __restrict__ ntype,
                                const float* __restrict__ skipp_l, // [T]
                                const float* __restrict__ lng_l,   // [T,64]
                                const float* __restrict__ lnb_l,   // [T,64]
                                float* __restrict__ out, int n) {
  int nd = blockIdx.x * blockDim.x + threadIdx.x;
  if (nd >= n) return;
  const int t = ntype[nd];
  const float alpha = sigmoidf_(skipp_l[t]);
  const float* tp = trans + (size_t)nd * DD;
  const float* xp = x + (size_t)nd * DD;
  float o[DD];
  float mu = 0.0f;
#pragma unroll
  for (int i = 0; i < DD; ++i) {
    o[i] = tp[i] * alpha + xp[i] * (1.0f - alpha);
    mu += o[i];
  }
  mu *= (1.0f / DD);
  float var = 0.0f;
#pragma unroll
  for (int i = 0; i < DD; ++i) {
    float dlt = o[i] - mu;
    var += dlt * dlt;
  }
  var *= (1.0f / DD);
  const float inv = rsqrtf(var + 1e-5f);
  float* op = out + (size_t)nd * DD;
#pragma unroll
  for (int i = 0; i < DD; ++i)
    op[i] = (o[i] - mu) * inv * lng_l[t * DD + i] + lnb_l[t * DD + i];
}

// ---------------------------------------------------------------------------
// Host orchestration
// ---------------------------------------------------------------------------
static inline int ceil_div(int a, int b) { return (a + b - 1) / b; }

extern "C" void kernel_launch(void* const* d_in, const int* in_sizes, int n_in,
                              void* d_out, int out_size, void* d_ws, size_t ws_size,
                              hipStream_t stream) {
  const float* h_in    = (const float*)d_in[0];
  const int*   src     = (const int*)  d_in[1];
  const int*   dst     = (const int*)  d_in[2];
  const int*   etype   = (const int*)  d_in[3];
  const int*   ntype   = (const int*)  d_in[4];
  const float* adapt_W = (const float*)d_in[5];
  const float* adapt_b = (const float*)d_in[6];
  const float* Wk      = (const float*)d_in[7];
  const float* bk      = (const float*)d_in[8];
  const float* Wq      = (const float*)d_in[9];
  const float* bq      = (const float*)d_in[10];
  const float* Wv      = (const float*)d_in[11];
  const float* bv      = (const float*)d_in[12];
  const float* Wa      = (const float*)d_in[13];
  const float* ba      = (const float*)d_in[14];
  const float* ln_g    = (const float*)d_in[15];
  const float* ln_b    = (const float*)d_in[16];
  const float* rel_att = (const float*)d_in[17];
  const float* rel_msg = (const float*)d_in[18];
  const float* rel_pri = (const float*)d_in[19];
  const float* nta     = (const float*)d_in[20];
  const float* nta1    = (const float*)d_in[21];
  const float* skipp   = (const float*)d_in[22];
  const float* wgt     = (const float*)d_in[23];
  const float* afc_w   = (const float*)d_in[24];

  const int N = in_sizes[4];
  const int E = in_sizes[1];

  // Workspace layout (floats, then f16 weight shadows)
  float* ws = (float*)d_ws;
  const size_t nf = (size_t)N * DD;
  float* xA     = ws;
  float* xB     = xA + nf;
  float* kb     = xB + nf;
  float* qb     = kb + nf;
  float* vb     = qb + nf;
  float* acc    = vb + nf;
  float* trans  = acc + nf;
  float* logits = trans + nf;                     // E*H
  float* segmax = logits + (size_t)E * HH;        // N*R*H
  float* den    = segmax + (size_t)N * RR * HH;   // N*R*H
  float* cnt    = den + (size_t)N * RR * HH;      // N*R
  _Float16* wf16 = (_Float16*)(cnt + (size_t)N * RR);
  _Float16* adWt = wf16;                          // [T,64,128]
  _Float16* WkT  = adWt + (size_t)TT * DD * D_IN; // [L*T,64,64] each
  _Float16* WqT  = WkT + (size_t)LL * TT * DD * DD;
  _Float16* WvT  = WqT + (size_t)LL * TT * DD * DD;
  _Float16* WaT  = WvT + (size_t)LL * TT * DD * DD;

  const dim3 blk(256);
  const int proj_grid = ceil_div(N, 16 * 8);      // 8 waves/block, 16 nodes/wave
  const dim3 egrid(ceil_div(E, 256), HH);
  const int ehgrid = ceil_div(E * HH, 256);
  const int ngrid  = ceil_div(N, 256);

  // Pre-convert weights to f16, transposed [t][col][K]
  {
    int ta = TT * D_IN * DD;
    convert_w_kernel<<<ceil_div(ta, 256), blk, 0, stream>>>(adapt_W, adWt, TT, D_IN);
    int tw = LL * TT * DD * DD;
    convert_w_kernel<<<ceil_div(tw, 256), blk, 0, stream>>>(Wk, WkT, LL * TT, DD);
    convert_w_kernel<<<ceil_div(tw, 256), blk, 0, stream>>>(Wq, WqT, LL * TT, DD);
    convert_w_kernel<<<ceil_div(tw, 256), blk, 0, stream>>>(Wv, WvT, LL * TT, DD);
    convert_w_kernel<<<ceil_div(tw, 256), blk, 0, stream>>>(Wa, WaT, LL * TT, DD);
  }

  // adapt: x = tanh(per_type_linear(h, adapt_W, adapt_b))
  proj_kernel<D_IN, 1><<<proj_grid, blk, 0, stream>>>(
      h_in, adWt, adapt_b, ntype, xA, N);

  float* xcur = xA;
  float* xnext = xB;
  for (int l = 0; l < LL; ++l) {
    const size_t wOff  = (size_t)l * TT * DD * DD;
    const size_t bOff  = (size_t)l * TT * DD;
    const size_t rOff  = (size_t)l * RR * HH * DK * DK;
    const size_t pOff  = (size_t)l * RR * HH;
    const size_t tOff  = (size_t)l * TT;

    // k / q / v projections (WMMA)
    proj_kernel<DD, 0><<<proj_grid, blk, 0, stream>>>(
        xcur, WkT + wOff, bk + bOff, ntype, kb, N);
    proj_kernel<DD, 0><<<proj_grid, blk, 0, stream>>>(
        xcur, WqT + wOff, bq + bOff, ntype, qb, N);
    proj_kernel<DD, 0><<<proj_grid, blk, 0, stream>>>(
        xcur, WvT + wOff, bv + bOff, ntype, vb, N);

    // init segment scratch
    {
      size_t ns = (size_t)N * RR * HH;
      fill_kernel<<<(unsigned)((ns + 255) / 256), blk, 0, stream>>>(segmax, -INFINITY, ns);
      fill_kernel<<<(unsigned)((ns + 255) / 256), blk, 0, stream>>>(den, 0.0f, ns);
      size_t nc = (size_t)N * RR;
      fill_kernel<<<(unsigned)((nc + 255) / 256), blk, 0, stream>>>(cnt, 0.0f, nc);
      fill_kernel<<<(unsigned)((nf + 255) / 256), blk, 0, stream>>>(acc, 0.0f, nf);
    }

    // edge passes
    edge_logits_kernel<<<egrid, blk, 0, stream>>>(
        src, dst, etype, kb, qb,
        rel_att + rOff, rel_pri + pOff,
        nta + tOff, nta1 + tOff, afc_w + (size_t)l * 2 * DK, wgt + l,
        logits, segmax, cnt, E);
    edge_exp_kernel<<<ehgrid, blk, 0, stream>>>(dst, etype, logits, segmax, den, E);
    edge_accum_kernel<<<egrid, blk, 0, stream>>>(
        src, dst, etype, vb, rel_msg + rOff, logits, den, acc, E);

    // cross-etype mean
    node_div_kernel<<<ngrid, blk, 0, stream>>>(acc, cnt, N);

    // output projection (WMMA) + epilogue
    proj_kernel<DD, 0><<<proj_grid, blk, 0, stream>>>(
        acc, WaT + wOff, ba + bOff, ntype, trans, N);

    float* dest = (l == LL - 1) ? (float*)d_out : xnext;
    epilogue_kernel<<<ngrid, blk, 0, stream>>>(
        trans, xcur, ntype, skipp + tOff,
        ln_g + (size_t)l * TT * DD, ln_b + (size_t)l * TT * DD, dest, N);

    if (l < LL - 1) { float* tmp = xcur; xcur = xnext; xnext = tmp; }
  }
}
